// SaccLayer_54614804136359
// MI455X (gfx1250) — compile-verified, matrix operand order checked by reading, not run on hardware
//
#include <hip/hip_runtime.h>

// ---------------------------------------------------------------------------
// CDNA5 / gfx1250 implementation of the SaccLayer reference.
// Heavy GEMMs: v_wmma_f32_16x16x32_bf16, bf16 pre-converted operands,
// async global->LDS double-buffered B-tile staging (ASYNCcnt pipelining),
// 32Mx64N wave tiles (8 WMMAs per k-step, B-fragment register reuse).
// ---------------------------------------------------------------------------

#define DIM   768
#define NH    12
#define HD    64
#define BS_   128
#define WS_   64
#define NSAC  3
#define BQ    4
#define NN    8192
#define NB    64                 // NN / BS_
#define XO_N  (BQ * NN * DIM)    // 25165824

typedef unsigned short u16;
typedef __attribute__((ext_vector_type(16))) __bf16        v16bf;
typedef __attribute__((ext_vector_type(8)))  float         v8f;
typedef __attribute__((ext_vector_type(8)))  unsigned int  v8u;

__device__ __forceinline__ unsigned int f2bf(float f) {
  unsigned int u = __float_as_uint(f);
  return (u + 0x7FFFu + ((u >> 16) & 1u)) >> 16;   // round-to-nearest-even
}

__device__ __forceinline__ float geluf(float x) {
  return 0.5f * x * (1.0f + erff(x * 0.70710678118654752f));
}

// ---------------------------------------------------------------------------
// f32 -> bf16 conversion (weights once per launch; small activations per use)
// ---------------------------------------------------------------------------
__global__ void k_cvt_bf16(const float* __restrict__ s, u16* __restrict__ d, int n) {
  const int i = blockIdx.x * 256 + threadIdx.x;
  if (i < n) d[i] = (u16)f2bf(s[i]);
}

// ---------------------------------------------------------------------------
// WMMA GEMM:  out[M,N] = act(A[M,K] @ W[N,K]^T + bias[N])
//   A, W bf16. Output: f32 store / f32 accumulate (do_acc) or bf16 (Cb).
// Block: 128 thr = 4 waves. Block tile 128Mx64N, wave tile 32Mx64N.
// B tile (64 rows x 32 k = 4KB) staged in LDS (double-buffered) via
// global_load_async_to_lds_b128 + s_wait_asynccnt. Each B fragment feeds two
// WMMAs (M=+0 / M=+16) -> 8 WMMAs per 32-wide k-step per wave.
// N must be a multiple of 64; M arbitrary (clamped loads keep EXEC full).
// ---------------------------------------------------------------------------
__global__ void __launch_bounds__(128)
k_gemm(const u16* __restrict__ A, int lda, int M,
       const u16* __restrict__ Wt, int ldw, int N,
       const float* __restrict__ bias,
       float* __restrict__ C, u16* __restrict__ Cb, int ldc,
       int K, int do_gelu, int do_acc) {
  __shared__ unsigned char tb[2][64 * 64];   // [buf][row*64B]: 64 rows x 32 bf16
  const int tid  = threadIdx.x;
  const int lane = tid & 31;
  const int wave = tid >> 5;
  const int m0 = blockIdx.y * 128 + wave * 32;
  const int n0 = blockIdx.x * 64;

  int ar0 = m0 + (lane & 15);       if (ar0 >= M) ar0 = M - 1;
  int ar1 = m0 + 16 + (lane & 15);  if (ar1 >= M) ar1 = M - 1;
  const int koff = (lane >> 4) << 4;         // lanes 16-31 take K+16..31
  const u16* ap0 = A + (size_t)ar0 * lda + koff;
  const u16* ap1 = A + (size_t)ar1 * lda + koff;

  // Staging assignment: 256 16-byte chunks per 4KB tile, 2 per thread.
  const int r0 = tid >> 2,          s0 = (tid & 3) * 16;
  const int r1 = (tid + 128) >> 2,  s1 = ((tid + 128) & 3) * 16;
  const u16* g0 = Wt + (size_t)(n0 + r0) * ldw + (s0 >> 1);
  const u16* g1 = Wt + (size_t)(n0 + r1) * ldw + (s1 >> 1);
  const unsigned l0 = (unsigned)(size_t)&tb[0][r0 * 64 + s0];
  const unsigned l1 = (unsigned)(size_t)&tb[0][r1 * 64 + s1];

  // Prologue: stage k-step 0 into buffer 0.
  asm volatile("global_load_async_to_lds_b128 %0, %1, off" :: "v"(l0), "v"(g0) : "memory");
  asm volatile("global_load_async_to_lds_b128 %0, %1, off" :: "v"(l1), "v"(g1) : "memory");

  v8f acc[8] = {};
  const int nk = K >> 5;
  for (int i = 0; i < nk; ++i) {
    asm volatile("s_wait_asynccnt 0" ::: "memory");   // my async writes done
    __syncthreads();                                  // everyone's writes done
    if (i + 1 < nk) {                                 // prefetch next B tile
      const int kb = (i + 1) << 5;
      const unsigned lb = ((i + 1) & 1) ? 4096u : 0u;
      asm volatile("global_load_async_to_lds_b128 %0, %1, off"
                   :: "v"(l0 + lb), "v"(g0 + kb) : "memory");
      asm volatile("global_load_async_to_lds_b128 %0, %1, off"
                   :: "v"(l1 + lb), "v"(g1 + kb) : "memory");
    }
    const unsigned char* bbase = &tb[i & 1][0];
    const v16bf a0 = __builtin_bit_cast(v16bf, *(const v8u*)(ap0 + (i << 5)));
    const v16bf a1 = __builtin_bit_cast(v16bf, *(const v8u*)(ap1 + (i << 5)));
    v8u br[4];
#pragma unroll
    for (int j = 0; j < 4; ++j)
      br[j] = *(const v8u*)(bbase + (j * 16 + (lane & 15)) * 64 + (lane >> 4) * 32);
#pragma unroll
    for (int j = 0; j < 4; ++j) {
      const v16bf b = __builtin_bit_cast(v16bf, br[j]);
      acc[2 * j] = __builtin_amdgcn_wmma_f32_16x16x32_bf16(
          false, a0, false, b, (short)0, acc[2 * j], false, false);
      acc[2 * j + 1] = __builtin_amdgcn_wmma_f32_16x16x32_bf16(
          false, a1, false, b, (short)0, acc[2 * j + 1], false, false);
    }
    __syncthreads();                                  // done reading this buffer
  }

  // Epilogue: VGPR r of lane L holds (m = mbase + r + 8*(L>=16), n = n0+j*16+(L&15))
  const int nc = lane & 15;
  const int mhi = 8 * (lane >> 4);
#pragma unroll
  for (int j = 0; j < 4; ++j) {
    const int n = n0 + j * 16 + nc;
    const float bv = bias[n];
#pragma unroll
    for (int mi = 0; mi < 2; ++mi) {
      const int mbase = m0 + mi * 16 + mhi;
#pragma unroll
      for (int r = 0; r < 8; ++r) {
        const int m = mbase + r;
        if (m < M) {
          float v = acc[2 * j + mi][r] + bv;
          if (do_gelu) v = geluf(v);
          if (Cb) {
            Cb[(size_t)m * ldc + n] = (u16)f2bf(v);
          } else {
            float* p = C + (size_t)m * ldc + n;
            if (do_acc) v += *p;
            *p = v;
          }
        }
      }
    }
  }
}

// ---------------------------------------------------------------------------
// LayerNorm over last dim (=768); writes f32 (y32) and/or bf16 (y16).
// ---------------------------------------------------------------------------
__global__ void k_ln(const float* __restrict__ x, const float* __restrict__ g,
                     const float* __restrict__ b, float* __restrict__ y32,
                     u16* __restrict__ y16, int M) {
  const int row = blockIdx.x;
  if (row >= M) return;
  const float* xr = x + (size_t)row * DIM;
  __shared__ float red[256];
  const int tid = threadIdx.x;

  float s = 0.f;
  for (int d = tid; d < DIM; d += 256) s += xr[d];
  red[tid] = s; __syncthreads();
  for (int k = 128; k > 0; k >>= 1) { if (tid < k) red[tid] += red[tid + k]; __syncthreads(); }
  const float mean = red[0] * (1.0f / DIM);
  __syncthreads();

  float v = 0.f;
  for (int d = tid; d < DIM; d += 256) { float t = xr[d] - mean; v += t * t; }
  red[tid] = v; __syncthreads();
  for (int k = 128; k > 0; k >>= 1) { if (tid < k) red[tid] += red[tid + k]; __syncthreads(); }
  const float rstd = rsqrtf(red[0] * (1.0f / DIM) + 1e-5f);

  for (int d = tid; d < DIM; d += 256) {
    const float o = (xr[d] - mean) * rstd * g[d] + b[d];
    if (y32) y32[(size_t)row * DIM + d] = o;
    if (y16) y16[(size_t)row * DIM + d] = (u16)f2bf(o);
  }
}

// ---------------------------------------------------------------------------
// Block pooling: softmax-weighted mean/std + max, 3D->D proj, LN, +pos.
// ---------------------------------------------------------------------------
__global__ void k_block_pool(const float* __restrict__ H_,
                             const float* __restrict__ pw, const float* __restrict__ pwb,
                             const float* __restrict__ projw, const float* __restrict__ projb,
                             const float* __restrict__ ng, const float* __restrict__ nb_,
                             const float* __restrict__ pos, float* __restrict__ pm) {
  const int blk = blockIdx.x & 63;
  const int b = blockIdx.x >> 6;
  const float* base = H_ + ((size_t)b * NN + (size_t)blk * BS_) * DIM;
  __shared__ float lg[BS_];
  __shared__ float feat[3 * DIM];
  __shared__ float outv[DIM];
  __shared__ float red[256];
  const int tid = threadIdx.x;

  if (tid < BS_) {
    const float* r = base + (size_t)tid * DIM;
    float s = 0.f;
    for (int d = 0; d < DIM; ++d) s += r[d] * pw[d];
    lg[tid] = s + pwb[0];
  }
  __syncthreads();
  if (tid == 0) {
    float mx = lg[0];
    for (int i = 1; i < BS_; ++i) mx = fmaxf(mx, lg[i]);
    float ssum = 0.f;
    for (int i = 0; i < BS_; ++i) { lg[i] = expf(lg[i] - mx); ssum += lg[i]; }
    const float inv = 1.f / ssum;
    for (int i = 0; i < BS_; ++i) lg[i] *= inv;
  }
  __syncthreads();

  for (int d = tid; d < DIM; d += 256) {
    float mn = 0.f, mx = -3.4e38f;
    for (int r = 0; r < BS_; ++r) {
      const float v = base[(size_t)r * DIM + d];
      mn += lg[r] * v; mx = fmaxf(mx, v);
    }
    float var = 0.f;
    for (int r = 0; r < BS_; ++r) {
      const float v = base[(size_t)r * DIM + d] - mn;
      var += lg[r] * v * v;
    }
    feat[d] = mn; feat[DIM + d] = sqrtf(var + 1e-8f); feat[2 * DIM + d] = mx;
  }
  __syncthreads();

  for (int d = tid; d < DIM; d += 256) {
    const float* wr = projw + (size_t)d * (3 * DIM);
    float s = projb[d];
    for (int k = 0; k < 3 * DIM; ++k) s += feat[k] * wr[k];
    outv[d] = s;
  }
  __syncthreads();

  float s2 = 0.f;
  for (int d = tid; d < DIM; d += 256) s2 += outv[d];
  red[tid] = s2; __syncthreads();
  for (int k = 128; k > 0; k >>= 1) { if (tid < k) red[tid] += red[tid + k]; __syncthreads(); }
  const float mean = red[0] * (1.0f / DIM);
  __syncthreads();
  float v2 = 0.f;
  for (int d = tid; d < DIM; d += 256) { float t = outv[d] - mean; v2 += t * t; }
  red[tid] = v2; __syncthreads();
  for (int k = 128; k > 0; k >>= 1) { if (tid < k) red[tid] += red[tid + k]; __syncthreads(); }
  const float rstd = rsqrtf(red[0] * (1.0f / DIM) + 1e-5f);

  float* dst = pm + ((size_t)b * NB + blk) * DIM;
  for (int d = tid; d < DIM; d += 256)
    dst[d] = (outv[d] - mean) * rstd * ng[d] + nb_[d] + pos[(size_t)blk * DIM + d];
}

__global__ void k_st_mean(const float* __restrict__ pm, float* __restrict__ st) {
  const int b = blockIdx.x, tid = threadIdx.x;
  for (int d = tid; d < DIM; d += 256) {
    float s = 0.f;
    for (int k = 0; k < NB; ++k) s += pm[((size_t)b * NB + k) * DIM + d];
    st[b * DIM + d] = s * (1.0f / NB);
  }
}

__global__ void k_logits(const float* __restrict__ q, const float* __restrict__ kp,
                         float* __restrict__ lg, float* __restrict__ fls) {
  const int m = blockIdx.x & 63, b = blockIdx.x >> 6, tid = threadIdx.x;
  __shared__ float red[256];
  const float* qv = q + (size_t)b * DIM;
  const float* kv = kp + ((size_t)b * NB + m) * DIM;
  float s = 0.f;
  for (int d = tid; d < DIM; d += 256) s += qv[d] * kv[d];
  red[tid] = s; __syncthreads();
  for (int k = 128; k > 0; k >>= 1) { if (tid < k) red[tid] += red[tid + k]; __syncthreads(); }
  if (tid == 0) {
    const float v = red[0] * 0.03608439182435161f;   // 1/sqrt(768)
    lg[b * NB + m] = v;
    fls[b * NB + m] = v;
  }
}

__global__ void k_argmax(const float* __restrict__ lg, float* __restrict__ fps,
                         int* __restrict__ starts) {
  const int b = blockIdx.x;
  if (threadIdx.x == 0) {
    const float* l = lg + b * NB;
    int best = 0; float bv = l[0];
    for (int i = 1; i < NB; ++i) if (l[i] > bv) { bv = l[i]; best = i; }
    const int fp = best * BS_;
    fps[b] = (float)fp;
    int st = fp - WS_ / 2;
    if (st < 0) st = 0;
    if (st > NN - WS_) st = NN - WS_;
    starts[b] = st;
  }
}

// ctx = [st, ac(0..t*64), win]; appends win into ac (f32 + bf16 copies).
__global__ void k_build_ctx(int t, const float* __restrict__ st, float* __restrict__ ac,
                            u16* __restrict__ acb, const float* __restrict__ h,
                            const int* __restrict__ starts, float* __restrict__ ctx, int Lc) {
  const int i = blockIdx.x, b = blockIdx.y, tid = threadIdx.x;
  float* dst = ctx + ((size_t)b * Lc + i) * DIM;
  if (i == 0) {
    for (int d = tid; d < DIM; d += 256) dst[d] = st[b * DIM + d];
  } else if (i <= t * WS_) {
    const float* s = ac + ((size_t)b * (NSAC * WS_) + (i - 1)) * DIM;
    for (int d = tid; d < DIM; d += 256) dst[d] = s[d];
  } else {
    const int j = i - 1 - t * WS_;
    const int s0 = starts[b];
    const float* s = h + ((size_t)b * NN + s0 + j) * DIM;
    const size_t arow = (size_t)b * (NSAC * WS_) + t * WS_ + j;
    float* acd = ac + arow * DIM;
    u16* acd16 = acb + arow * DIM;
    for (int d = tid; d < DIM; d += 256) {
      const float v = s[d];
      dst[d] = v; acd[d] = v; acd16[d] = (u16)f2bf(v);
    }
  }
}

// Per-(b,h,query) softmax attention; Lk <= 193. Output bf16 (feeds GEMM).
__global__ void k_attn(const float* __restrict__ Qb, int q_bs, int q_rs, int q_off,
                       const float* __restrict__ Kb, int k_bs, int k_rs, int k_off,
                       const float* __restrict__ Vb, int v_bs, int v_rs, int v_off,
                       u16* __restrict__ Ob, int o_bs, int o_rs, int Lk) {
  const int qi = blockIdx.x, h = blockIdx.y, b = blockIdx.z, tid = threadIdx.x;
  __shared__ float qv[HD];
  __shared__ float sc[224];
  __shared__ float stats[1];
  const float* qp = Qb + (size_t)b * q_bs + (size_t)qi * q_rs + q_off + h * HD;
  if (tid < HD) qv[tid] = qp[tid];
  __syncthreads();
  for (int j = tid; j < Lk; j += 64) {
    const float* kp = Kb + (size_t)b * k_bs + (size_t)j * k_rs + k_off + h * HD;
    float s = 0.f;
    for (int d = 0; d < HD; ++d) s += qv[d] * kp[d];
    sc[j] = s * 0.125f;                       // 1/sqrt(64)
  }
  __syncthreads();
  if (tid == 0) {
    float mx = sc[0];
    for (int j = 1; j < Lk; ++j) mx = fmaxf(mx, sc[j]);
    float ssum = 0.f;
    for (int j = 0; j < Lk; ++j) { sc[j] = expf(sc[j] - mx); ssum += sc[j]; }
    stats[0] = 1.f / ssum;
  }
  __syncthreads();
  const float rinv = stats[0];
  float o = 0.f;
  for (int j = 0; j < Lk; ++j) {
    const float* vp = Vb + (size_t)b * v_bs + (size_t)j * v_rs + v_off + h * HD;
    o += sc[j] * vp[tid];
  }
  Ob[(size_t)b * o_bs + (size_t)qi * o_rs + h * HD + tid] = (u16)f2bf(o * rinv);
}

__global__ void k_extract_cl(const float* __restrict__ ctx, int Lc,
                             float* __restrict__ cl, float* __restrict__ st) {
  const int b = blockIdx.x, tid = threadIdx.x;
  for (int d = tid; d < DIM; d += 256) {
    const float v = ctx[((size_t)b * Lc) * DIM + d];
    cl[b * DIM + d] = v;
    st[b * DIM + d] = v;
  }
}

__global__ void k_gate(const float* __restrict__ w1, const float* __restrict__ b1,
                       const float* __restrict__ w2, const float* __restrict__ b2,
                       float a_in, float* __restrict__ gate) {
  __shared__ float red[256];
  const int tid = threadIdx.x;
  float s = 0.f;
  for (int d = tid; d < DIM; d += 256) {
    const float hdn = geluf(a_in * w1[d] + b1[d]);
    s += hdn * w2[d];
  }
  red[tid] = s; __syncthreads();
  for (int k = 128; k > 0; k >>= 1) { if (tid < k) red[tid] += red[tid + k]; __syncthreads(); }
  if (tid == 0) gate[0] = 1.f / (1.f + expf(-(red[0] + b2[0])));
}

__global__ void k_pm_update(float* __restrict__ pm, const float* __restrict__ d,
                            const float* __restrict__ gate) {
  const size_t i = (size_t)blockIdx.x * 256 + threadIdx.x;
  if (i < (size_t)BQ * NB * DIM) pm[i] += gate[0] * d[i];
}

__global__ void k_xo(const float* __restrict__ x, const float* __restrict__ u,
                     float* __restrict__ out) {
  const size_t i = (size_t)blockIdx.x * 256 + threadIdx.x;
  const int b = (int)(i / ((size_t)NN * DIM));
  const int d = (int)(i % DIM);
  out[i] = x[i] + u[b * DIM + d];
}

// ---------------------------------------------------------------------------
// Host-side orchestration
// ---------------------------------------------------------------------------
extern "C" void kernel_launch(void* const* d_in, const int* in_sizes, int n_in,
                              void* d_out, int out_size, void* d_ws, size_t ws_size,
                              hipStream_t stream) {
  (void)in_sizes; (void)n_in; (void)out_size; (void)ws_size;
  const float* x          = (const float*)d_in[0];
  const float* pe_w_w     = (const float*)d_in[1];
  const float* pe_w_b     = (const float*)d_in[2];
  const float* pe_proj_w  = (const float*)d_in[3];
  const float* pe_proj_b  = (const float*)d_in[4];
  const float* pe_norm_g  = (const float*)d_in[5];
  const float* pe_norm_b  = (const float*)d_in[6];
  const float* pe_pos     = (const float*)d_in[7];
  const float* qp_w       = (const float*)d_in[8];
  const float* qp_b       = (const float*)d_in[9];
  const float* kp_w       = (const float*)d_in[10];
  const float* kp_b       = (const float*)d_in[11];
  const float* fov_in_w   = (const float*)d_in[12];
  const float* fov_in_b   = (const float*)d_in[13];
  const float* fov_out_w  = (const float*)d_in[14];
  const float* fov_out_b  = (const float*)d_in[15];
  const float* fov_n1_g   = (const float*)d_in[16];
  const float* fov_n1_b   = (const float*)d_in[17];
  const float* fov_n2_g   = (const float*)d_in[18];
  const float* fov_n2_b   = (const float*)d_in[19];
  const float* fov_ff_w1  = (const float*)d_in[20];
  const float* fov_ff_b1  = (const float*)d_in[21];
  const float* fov_ff_w2  = (const float*)d_in[22];
  const float* fov_ff_b2  = (const float*)d_in[23];
  const float* ma_in_w    = (const float*)d_in[24];
  const float* ma_in_b    = (const float*)d_in[25];
  const float* ma_out_w   = (const float*)d_in[26];
  const float* ma_out_b   = (const float*)d_in[27];
  const float* mn_g       = (const float*)d_in[28];
  const float* mn_b       = (const float*)d_in[29];
  const float* mg_w1      = (const float*)d_in[30];
  const float* mg_b1      = (const float*)d_in[31];
  const float* mg_w2      = (const float*)d_in[32];
  const float* mg_b2      = (const float*)d_in[33];
  const float* op_w       = (const float*)d_in[34];
  const float* op_b       = (const float*)d_in[35];
  const float* on_g       = (const float*)d_in[36];
  const float* on_b       = (const float*)d_in[37];
  const float* l1_g       = (const float*)d_in[38];
  const float* l1_b       = (const float*)d_in[39];
  const float* l2_g       = (const float*)d_in[40];
  const float* l2_b       = (const float*)d_in[41];
  const float* mlp_w1     = (const float*)d_in[42];
  const float* mlp_b1     = (const float*)d_in[43];
  const float* mlp_w2     = (const float*)d_in[44];
  const float* mlp_b2     = (const float*)d_in[45];

  float* out = (float*)d_out;
  float* xo  = out;                     // (B,N,D)
  float* fps = out + XO_N;              // (3,4)
  float* fls = out + XO_N + NSAC * BQ;  // (3,4,64)

  // Workspace carve-out (32B-aligned slabs)
  float* W = (float*)d_ws;
  size_t off = 0;
  auto alloc = [&](size_t nf) {
    nf = (nf + 15) & ~(size_t)15;
    float* p = W + off; off += nf; return p;
  };
  auto alloc16 = [&](size_t ne) { return (u16*)alloc((ne + 1) / 2); };

  float* H     = alloc((size_t)XO_N);                    // LN(x), f32 (pool+gather)
  float* PM    = alloc((size_t)BQ * NB * DIM);
  u16*   PMb   = alloc16((size_t)BQ * NB * DIM);
  float* ST    = alloc((size_t)BQ * DIM);
  u16*   STb   = alloc16((size_t)BQ * DIM);
  u16*   LNBb  = alloc16((size_t)8192 * DIM);            // bf16 LN outputs (GEMM A)
  float* QKV   = alloc((size_t)BQ * 193 * 3 * DIM);
  float* CTX   = alloc((size_t)BQ * 193 * DIM);
  u16*   ATTb  = alloc16((size_t)BQ * 193 * DIM);
  float* AC    = alloc((size_t)BQ * NSAC * WS_ * DIM);
  u16*   ACb   = alloc16((size_t)BQ * NSAC * WS_ * DIM);
  float* KV    = alloc((size_t)BQ * NSAC * WS_ * 2 * DIM);
  float* QP    = alloc((size_t)BQ * NB * DIM);
  float* KPB   = alloc((size_t)BQ * NB * DIM);
  float* QV    = alloc((size_t)BQ * DIM);
  float* LG    = alloc(BQ * NB);
  float* DB    = alloc((size_t)BQ * NB * DIM);
  u16*   HIDb  = alloc16((size_t)8192 * 4 * DIM);        // bf16 GELU hidden
  float* CL    = alloc((size_t)BQ * DIM);
  float* U     = alloc((size_t)BQ * DIM);
  float* GATE  = alloc(16);
  int*   STARTS = (int*)alloc(16);
  // bf16 weight copies
  u16* Wqp    = alloc16((size_t)DIM * DIM);
  u16* Wkp    = alloc16((size_t)DIM * DIM);
  u16* Wfin   = alloc16((size_t)3 * DIM * DIM);
  u16* Wfout  = alloc16((size_t)DIM * DIM);
  u16* Wff1   = alloc16((size_t)4 * DIM * DIM);
  u16* Wff2   = alloc16((size_t)4 * DIM * DIM);
  u16* Wmain  = alloc16((size_t)3 * DIM * DIM);
  u16* Wmaout = alloc16((size_t)DIM * DIM);
  u16* Wop    = alloc16((size_t)DIM * DIM);
  u16* Wm1    = alloc16((size_t)4 * DIM * DIM);
  u16* Wm2    = alloc16((size_t)4 * DIM * DIM);

  auto cvt = [&](const float* s, u16* d, size_t n) {
    k_cvt_bf16<<<((int)n + 255) / 256, 256, 0, stream>>>(s, d, (int)n);
  };
  auto gemm = [&](const u16* A, int lda, int M, const u16* Wm, int ldw, int N,
                  const float* bias, float* C, u16* Cb, int ldc, int K, int gel, int acc) {
    dim3 g(N / 64, (M + 127) / 128);
    k_gemm<<<g, 128, 0, stream>>>(A, lda, M, Wm, ldw, N, bias, C, Cb, ldc, K, gel, acc);
  };

  // 0. weight precision conversion (bf16 operands for WMMA)
  cvt(qp_w, Wqp, (size_t)DIM * DIM);
  cvt(kp_w, Wkp, (size_t)DIM * DIM);
  cvt(fov_in_w, Wfin, (size_t)3 * DIM * DIM);
  cvt(fov_out_w, Wfout, (size_t)DIM * DIM);
  cvt(fov_ff_w1, Wff1, (size_t)4 * DIM * DIM);
  cvt(fov_ff_w2, Wff2, (size_t)4 * DIM * DIM);
  cvt(ma_in_w, Wmain, (size_t)3 * DIM * DIM);
  cvt(ma_out_w, Wmaout, (size_t)DIM * DIM);
  cvt(op_w, Wop, (size_t)DIM * DIM);
  cvt(mlp_w1, Wm1, (size_t)4 * DIM * DIM);
  cvt(mlp_w2, Wm2, (size_t)4 * DIM * DIM);

  // 1. h = LN(x)   2. block pooling -> pm   3. st = mean(pm)
  k_ln<<<BQ * NN, 256, 0, stream>>>(x, l1_g, l1_b, H, nullptr, BQ * NN);
  k_block_pool<<<BQ * NB, 256, 0, stream>>>(H, pe_w_w, pe_w_b, pe_proj_w, pe_proj_b,
                                            pe_norm_g, pe_norm_b, pe_pos, PM);
  k_st_mean<<<BQ, 256, 0, stream>>>(PM, ST);

  // 4. saccade loop
  for (int t = 0; t < NSAC; ++t) {
    const int Lc = 1 + t * WS_ + WS_;
    const int La = WS_ * (t + 1);

    cvt(ST, STb, (size_t)BQ * DIM);
    cvt(PM, PMb, (size_t)BQ * NB * DIM);
    gemm(STb, DIM, BQ, Wqp, DIM, DIM, qp_b, QV, nullptr, DIM, DIM, 0, 0);
    gemm(PMb, DIM, BQ * NB, Wkp, DIM, DIM, kp_b, KPB, nullptr, DIM, DIM, 0, 0);
    k_logits<<<BQ * NB, 256, 0, stream>>>(QV, KPB, LG, fls + t * BQ * NB);
    k_argmax<<<BQ, 64, 0, stream>>>(LG, fps + t * BQ, STARTS);
    k_build_ctx<<<dim3(Lc, BQ), 256, 0, stream>>>(t, ST, AC, ACb, H, STARTS, CTX, Lc);

    // fov self-attention block
    k_ln<<<BQ * Lc, 256, 0, stream>>>(CTX, fov_n1_g, fov_n1_b, nullptr, LNBb, BQ * Lc);
    gemm(LNBb, DIM, BQ * Lc, Wfin, DIM, 3 * DIM, fov_in_b, QKV, nullptr, 3 * DIM, DIM, 0, 0);
    k_attn<<<dim3(Lc, NH, BQ), 64, 0, stream>>>(
        QKV, Lc * 3 * DIM, 3 * DIM, 0,
        QKV, Lc * 3 * DIM, 3 * DIM, DIM,
        QKV, Lc * 3 * DIM, 3 * DIM, 2 * DIM,
        ATTb, Lc * DIM, DIM, Lc);
    gemm(ATTb, DIM, BQ * Lc, Wfout, DIM, DIM, fov_out_b, CTX, nullptr, DIM, DIM, 0, 1);

    // st = cl + ffn(LN(cl))
    k_extract_cl<<<BQ, 256, 0, stream>>>(CTX, Lc, CL, ST);
    k_ln<<<BQ, 256, 0, stream>>>(CL, fov_n2_g, fov_n2_b, nullptr, LNBb, BQ);
    gemm(LNBb, DIM, BQ, Wff1, DIM, 4 * DIM, fov_ff_b1, nullptr, HIDb, 4 * DIM, DIM, 1, 0);
    gemm(HIDb, 4 * DIM, BQ, Wff2, 4 * DIM, DIM, fov_ff_b2, ST, nullptr, DIM, 4 * DIM, 0, 1);

    // memory attention: d = MHA(LN(pm), ac, ac)
    k_ln<<<BQ * NB, 256, 0, stream>>>(PM, mn_g, mn_b, nullptr, LNBb, BQ * NB);
    gemm(LNBb, DIM, BQ * NB, Wmain, DIM, DIM, ma_in_b, QP, nullptr, DIM, DIM, 0, 0);
    for (int b = 0; b < BQ; ++b) {
      gemm(ACb + (size_t)b * NSAC * WS_ * DIM, DIM, La,
           Wmain + (size_t)DIM * DIM, DIM, 2 * DIM, ma_in_b + DIM,
           KV + (size_t)b * NSAC * WS_ * 2 * DIM, nullptr, 2 * DIM, DIM, 0, 0);
    }
    k_attn<<<dim3(NB, NH, BQ), 64, 0, stream>>>(
        QP, NB * DIM, DIM, 0,
        KV, NSAC * WS_ * 2 * DIM, 2 * DIM, 0,
        KV, NSAC * WS_ * 2 * DIM, 2 * DIM, DIM,
        ATTb, NB * DIM, DIM, La);
    gemm(ATTb, DIM, BQ * NB, Wmaout, DIM, DIM, ma_out_b, DB, nullptr, DIM, DIM, 0, 0);

    k_gate<<<1, 256, 0, stream>>>(mg_w1, mg_b1, mg_w2, mg_b2, (float)t / (float)NSAC, GATE);
    k_pm_update<<<(BQ * NB * DIM + 255) / 256, 256, 0, stream>>>(PM, DB, GATE);
  }

  // 5. xo = x + broadcast(LN(st) @ op_w^T + op_b)
  k_ln<<<BQ, 256, 0, stream>>>(ST, on_g, on_b, nullptr, LNBb, BQ);
  gemm(LNBb, DIM, BQ, Wop, DIM, DIM, op_b, U, nullptr, DIM, DIM, 0, 0);
  k_xo<<<XO_N / 256, 256, 0, stream>>>(x, U, xo);

  // 6. xo += mlp(LN(xo)) -- chunked; both GEMMs on the WMMA pipe
  const int CHUNK = 8192;
  for (int c = 0; c < (BQ * NN) / CHUNK; ++c) {
    float* xc = xo + (size_t)c * CHUNK * DIM;
    k_ln<<<CHUNK, 256, 0, stream>>>(xc, l2_g, l2_b, nullptr, LNBb, CHUNK);
    gemm(LNBb, DIM, CHUNK, Wm1, DIM, 4 * DIM, mlp_b1, nullptr, HIDb, 4 * DIM, DIM, 1, 0);
    gemm(HIDb, 4 * DIM, CHUNK, Wm2, 4 * DIM, DIM, mlp_b2, xc, nullptr, DIM, 4 * DIM, 0, 1);
  }
}
